// AttentionDecoder_40913858462121
// MI455X (gfx1250) — compile-verified, matrix operand order checked by reading, not run on hardware
//
#include <hip/hip_runtime.h>
#include <hip/hip_bf16.h>
#include <math.h>

// ---------------------------------------------------------------------------
// AttentionDecoder for MI455X (gfx1250): f16 WMMA GEMMs (32x64 wave tile,
// double-buffered fragments) + f16 streaming of the two 128MB tensors
// (enc_proj, encoder_output) that dominate HBM traffic.
// ---------------------------------------------------------------------------

typedef __attribute__((ext_vector_type(16))) _Float16 v16h;
typedef __attribute__((ext_vector_type(8)))  _Float16 v8h;
typedef __attribute__((ext_vector_type(8)))  float    v8f;

#define BB   128
#define TT   48
#define DIN  64
#define SS   2048
#define DENC 256
#define HH   256

union FragA { v16h v; v8h h[2]; };

#define WMMA_F16(Af, Bf, Cf) \
  __builtin_amdgcn_wmma_f32_16x16x32_f16(false, (Af).v, false, (Bf), (short)0, (Cf), false, false)

// exp/rcp-based transcendentals: lower to v_exp_f32 + v_rcp_f32 (hardware
// TRANS ops) instead of the branchy libm expansions. Saturate correctly.
__device__ __forceinline__ float fast_tanh(float x) {
  float e = __expf(2.0f * x);
  return 1.0f - 2.0f * __builtin_amdgcn_rcpf(e + 1.0f);
}
__device__ __forceinline__ float sigmf(float x) {
  return __builtin_amdgcn_rcpf(1.0f + __expf(-x));
}

// ------------------------- f32 -> f16 convert ------------------------------
__global__ __launch_bounds__(256) void f32_to_f16_kernel(const float* __restrict__ src,
                                                         _Float16* __restrict__ dst,
                                                         long n) {
  long i = (long)blockIdx.x * blockDim.x + threadIdx.x;
  if (i < n) dst[i] = (_Float16)src[i];
}

// ------------------- pack B [K,N] f32 -> WMMA fragment layout --------------
// dst chunk index = (kt*NT + nt)*32 + lane ; chunk = v16h (32 bytes).
// Per CDNA5 ISA 16-bit operand layout: half = lane>>4,
//   element i<8  -> k = kt*32 + half*8 + i
//   element i>=8 -> k = kt*32 + 16 + half*8 + (i-8)
// lane%16 selects column n within the 16-wide tile.
__global__ __launch_bounds__(256) void pack_b_kernel(const float* __restrict__ src,
                                                     _Float16* __restrict__ dst,
                                                     int K, int N) {
  int NT = N >> 4, KT = K >> 5;
  int total = KT * NT * 32;
  int idx = blockIdx.x * blockDim.x + threadIdx.x;
  if (idx >= total) return;
  int lane = idx & 31;
  int tile = idx >> 5;
  int nt = tile % NT;
  int kt = tile / NT;
  int half = lane >> 4;
  int n = nt * 16 + (lane & 15);
  v16h outv;
#pragma unroll
  for (int i = 0; i < 16; ++i) {
    int k = kt * 32 + ((i < 8) ? (half * 8 + i) : (16 + half * 8 + (i - 8)));
    outv[i] = (_Float16)src[(size_t)k * N + n];
  }
  *((v16h*)dst + (size_t)tile * 32 + lane) = outv;
}

// --------------------------- generic WMMA GEMM -----------------------------
// C[M,N] (f32 accum) = A[M,K](f16,row-major,lda) x Bpacked[K,N] (+bias[N])
// (+extra[M,ldextra] f32). Optional f32 and/or f16 outputs with stride ldc.
// One wave computes a 32x64 strip: 2 A fragments x 4 B fragments = 8 WMMAs
// per k-step, with the next k-step's fragments loaded before the MMAs issue
// (software pipeline -> non-zero loadcnt thresholds at the waits).
__global__ __launch_bounds__(256) void gemm_wmma_kernel(
    const _Float16* __restrict__ A, int lda,
    const _Float16* __restrict__ Bp,
    const float* __restrict__ bias,
    const float* __restrict__ extra, long ldextra,
    float* __restrict__ Cf32,
    _Float16* __restrict__ Cf16,
    int ldc, int M, int N, int K) {
  const int NT = N >> 4;
  const int KT = K >> 5;
  const int nStripCols = N >> 6;
  const int totalStrips = (M >> 5) * nStripCols;
  int strip = blockIdx.x * 8 + (threadIdx.x >> 5);
  if (strip >= totalStrips) return;
  int lane = threadIdx.x & 31;
  int mt = strip / nStripCols;
  int ns = strip - mt * nStripCols;
  int m0 = mt << 5;
  int nt0 = ns << 2;          // first of 4 16-col tiles
  int half = lane >> 4;

  v8f acc[2][4] = {};

  const _Float16* arow0 = A + (size_t)(m0 + (lane & 15)) * lda + half * 8;
  const _Float16* arow1 = arow0 + (size_t)16 * lda;
  const v16h* bptr = (const v16h*)Bp + ((size_t)nt0 << 5) + lane;
  const size_t bstep = (size_t)NT << 5;

  FragA a0c, a1c;
  v16h b0c, b1c, b2c, b3c;
  a0c.h[0] = *(const v8h*)(arow0); a0c.h[1] = *(const v8h*)(arow0 + 16);
  a1c.h[0] = *(const v8h*)(arow1); a1c.h[1] = *(const v8h*)(arow1 + 16);
  b0c = bptr[0]; b1c = bptr[32]; b2c = bptr[64]; b3c = bptr[96];

  for (int kt = 0; kt < KT - 1; ++kt) {
    arow0 += 32; arow1 += 32; bptr += bstep;
    FragA a0n, a1n;
    v16h b0n, b1n, b2n, b3n;
    a0n.h[0] = *(const v8h*)(arow0); a0n.h[1] = *(const v8h*)(arow0 + 16);
    a1n.h[0] = *(const v8h*)(arow1); a1n.h[1] = *(const v8h*)(arow1 + 16);
    b0n = bptr[0]; b1n = bptr[32]; b2n = bptr[64]; b3n = bptr[96];

    acc[0][0] = WMMA_F16(a0c, b0c, acc[0][0]);
    acc[0][1] = WMMA_F16(a0c, b1c, acc[0][1]);
    acc[0][2] = WMMA_F16(a0c, b2c, acc[0][2]);
    acc[0][3] = WMMA_F16(a0c, b3c, acc[0][3]);
    acc[1][0] = WMMA_F16(a1c, b0c, acc[1][0]);
    acc[1][1] = WMMA_F16(a1c, b1c, acc[1][1]);
    acc[1][2] = WMMA_F16(a1c, b2c, acc[1][2]);
    acc[1][3] = WMMA_F16(a1c, b3c, acc[1][3]);

    a0c = a0n; a1c = a1n;
    b0c = b0n; b1c = b1n; b2c = b2n; b3c = b3n;
  }
  acc[0][0] = WMMA_F16(a0c, b0c, acc[0][0]);
  acc[0][1] = WMMA_F16(a0c, b1c, acc[0][1]);
  acc[0][2] = WMMA_F16(a0c, b2c, acc[0][2]);
  acc[0][3] = WMMA_F16(a0c, b3c, acc[0][3]);
  acc[1][0] = WMMA_F16(a1c, b0c, acc[1][0]);
  acc[1][1] = WMMA_F16(a1c, b1c, acc[1][1]);
  acc[1][2] = WMMA_F16(a1c, b2c, acc[1][2]);
  acc[1][3] = WMMA_F16(a1c, b3c, acc[1][3]);

  // C layout (ISA): element v -> row base + half*8 + v, col tile base + lane%16
  int rb = m0 + half * 8;
  int c0 = (nt0 << 4) + (lane & 15);
#pragma unroll
  for (int mh = 0; mh < 2; ++mh) {
#pragma unroll
    for (int j = 0; j < 4; ++j) {
      int col = c0 + j * 16;
      float bv = bias ? bias[col] : 0.0f;
#pragma unroll
      for (int v = 0; v < 8; ++v) {
        int r = rb + mh * 16 + v;
        float val = acc[mh][j][v] + bv;
        if (extra) val += extra[(size_t)r * ldextra + col];
        if (Cf32) Cf32[(size_t)r * ldc + col] = val;
        if (Cf16) Cf16[(size_t)r * ldc + col] = (_Float16)val;
      }
    }
  }
}

// ------------------------------ score --------------------------------------
// One wave per (b,s): score = sum_h tanh(enc_proj[b,s,h] + dec_proj[b,h])*V[h]
__global__ __launch_bounds__(256) void score_kernel(const _Float16* __restrict__ encproj,
                                                    const float* __restrict__ decproj,
                                                    const float* __restrict__ V,
                                                    float* __restrict__ scores) {
  long w = ((long)blockIdx.x * blockDim.x + threadIdx.x) >> 5;
  int lane = threadIdx.x & 31;
  if (w >= (long)BB * SS) return;
  int b = (int)(w >> 11);   // / SS
  const _Float16* ep = encproj + w * HH + lane * 8;
  const float* dp = decproj + (size_t)b * HH + lane * 8;
  const float* vp = V + lane * 8;
  v8h e = *(const v8h*)ep;
  float acc = 0.0f;
#pragma unroll
  for (int i = 0; i < 8; ++i) acc += fast_tanh((float)e[i] + dp[i]) * vp[i];
#pragma unroll
  for (int off = 16; off > 0; off >>= 1) acc += __shfl_xor(acc, off, 32);
  if (lane == 0) scores[w] = acc;
}

// ------------------------------ softmax ------------------------------------
__global__ __launch_bounds__(256) void softmax_kernel(float* __restrict__ scores) {
  int b = blockIdx.x;
  float* sc = scores + (size_t)b * SS;
  __shared__ float red[8];
  int tid = threadIdx.x, lane = tid & 31, wv = tid >> 5;
  float m = -3.0e38f;
  float vals[SS / 256];
  int k = 0;
  for (int s = tid; s < SS; s += 256) { vals[k] = sc[s]; m = fmaxf(m, vals[k]); ++k; }
#pragma unroll
  for (int off = 16; off > 0; off >>= 1) m = fmaxf(m, __shfl_xor(m, off, 32));
  if (lane == 0) red[wv] = m;
  __syncthreads();
  float mm = red[0];
#pragma unroll
  for (int i = 1; i < 8; ++i) mm = fmaxf(mm, red[i]);
  __syncthreads();
  float sum = 0.0f;
  k = 0;
  for (int s = tid; s < SS; s += 256) { vals[k] = __expf(vals[k] - mm); sum += vals[k]; ++k; }
#pragma unroll
  for (int off = 16; off > 0; off >>= 1) sum += __shfl_xor(sum, off, 32);
  if (lane == 0) red[wv] = sum;
  __syncthreads();
  float tot = red[0];
#pragma unroll
  for (int i = 1; i < 8; ++i) tot += red[i];
  float inv = 1.0f / tot;
  k = 0;
  for (int s = tid; s < SS; s += 256) sc[s] = vals[k++] * inv;
}

// ------------------------------ context ------------------------------------
// context[b,d] = sum_s attn[b,s] * enc_f16[b,s,d]
// Thread layout: lane = d-group of 8 (v8h loads, 512B/wave fully coalesced),
// wave = s-slice; LDS partial reduction across the 8 slices.
__global__ __launch_bounds__(256) void context_kernel(const float* __restrict__ attn,
                                                      const _Float16* __restrict__ enc,
                                                      _Float16* __restrict__ ctx) {
  int b = blockIdx.x;
  int dg = threadIdx.x & 31;   // 32 groups of 8 contiguous d
  int sl = threadIdx.x >> 5;   // 8 s-slices
  const float* a = attn + (size_t)b * SS;
  const _Float16* e = enc + (size_t)b * SS * DENC + dg * 8;
  float acc[8] = {};
  for (int s = sl; s < SS; s += 8) {
    __builtin_prefetch((const void*)(e + (size_t)(s + 32) * DENC), 0, 0);
    float wgt = a[s];
    v8h ev = *(const v8h*)(e + (size_t)s * DENC);
#pragma unroll
    for (int i = 0; i < 8; ++i) acc[i] += wgt * (float)ev[i];
  }
  __shared__ float part[8][DENC];
#pragma unroll
  for (int i = 0; i < 8; ++i) part[sl][dg * 8 + i] = acc[i];
  __syncthreads();
  int d = threadIdx.x;
  float r = part[0][d];
#pragma unroll
  for (int i = 1; i < 8; ++i) r += part[i][d];
  ctx[(size_t)b * DENC + d] = (_Float16)r;
}

// --------------------------- LSTM pointwise --------------------------------
// gates [B,4H] order i,f,g,o ; updates c (f32), h (f16 into zh[:,H:2H]),
// and writes h to out[b,t,:] (f32).
__global__ __launch_bounds__(256) void lstm_kernel(const float* __restrict__ gates,
                                                   float* __restrict__ c,
                                                   _Float16* __restrict__ zh,
                                                   float* __restrict__ out,
                                                   int t) {
  int idx = blockIdx.x * 256 + threadIdx.x;   // B*H
  int b = idx >> 8;
  int h = idx & 255;
  const float* g = gates + (size_t)b * (4 * HH);
  float ig = sigmf(g[h]);
  float fg = sigmf(g[HH + h]);
  float gg = fast_tanh(g[2 * HH + h]);
  float og = sigmf(g[3 * HH + h]);
  float cn = fg * c[idx] + ig * gg;
  float hn = og * fast_tanh(cn);
  c[idx] = cn;
  zh[(size_t)b * (2 * HH) + HH + h] = (_Float16)hn;
  out[((size_t)b * TT + t) * HH + h] = hn;
}

// ------------------------------ init state ---------------------------------
__global__ __launch_bounds__(256) void init_state_kernel(const float* __restrict__ h0,
                                                         const float* __restrict__ c0,
                                                         _Float16* __restrict__ zh,
                                                         float* __restrict__ c) {
  int idx = blockIdx.x * 256 + threadIdx.x;   // B*H
  int b = idx >> 8;
  int h = idx & 255;
  zh[(size_t)b * (2 * HH) + HH + h] = (_Float16)h0[idx];
  c[idx] = c0[idx];
}

// ---------------------------------------------------------------------------
extern "C" void kernel_launch(void* const* d_in, const int* in_sizes, int n_in,
                              void* d_out, int out_size, void* d_ws, size_t ws_size,
                              hipStream_t stream) {
  (void)in_sizes; (void)n_in; (void)out_size; (void)ws_size;
  const float* inputs = (const float*)d_in[0];
  const float* h0     = (const float*)d_in[1];
  const float* c0     = (const float*)d_in[2];
  const float* enc    = (const float*)d_in[3];
  const float* W1     = (const float*)d_in[4];
  const float* b1     = (const float*)d_in[5];
  const float* W2     = (const float*)d_in[6];
  const float* b2     = (const float*)d_in[7];
  const float* V      = (const float*)d_in[8];
  // d_in[9] = bV : constant pre-softmax bias, cancels in softmax -> skipped.
  const float* Wp     = (const float*)d_in[10];
  const float* bp     = (const float*)d_in[11];
  const float* Wk     = (const float*)d_in[12];
  const float* Wr     = (const float*)d_in[13];
  const float* bl     = (const float*)d_in[14];
  float* out = (float*)d_out;

  // ---- carve workspace ----
  char* wptr = (char*)d_ws;
  auto take = [&](size_t bytes) {
    char* r = wptr;
    wptr += (bytes + 255) & ~(size_t)255;
    return (void*)r;
  };
  _Float16* encf16  = (_Float16*)take((size_t)BB * SS * DENC * 2);
  _Float16* encproj = (_Float16*)take((size_t)BB * SS * HH * 2);
  _Float16* W1p     = (_Float16*)take((size_t)DENC * HH * 2);
  _Float16* W2p     = (_Float16*)take((size_t)HH * HH * 2);
  _Float16* WpTopP  = (_Float16*)take((size_t)DIN * HH * 2);
  _Float16* WpBotP  = (_Float16*)take((size_t)DENC * HH * 2);
  _Float16* WkrP    = (_Float16*)take((size_t)(2 * HH) * (4 * HH) * 2);
  _Float16* xinf16  = (_Float16*)take((size_t)BB * TT * DIN * 2);
  float*    xproj   = (float*)take((size_t)BB * TT * HH * 4);
  float*    decproj = (float*)take((size_t)BB * HH * 4);
  float*    scores  = (float*)take((size_t)BB * SS * 4);
  _Float16* ctx     = (_Float16*)take((size_t)BB * DENC * 2);
  _Float16* zh      = (_Float16*)take((size_t)BB * 2 * HH * 2);
  float*    gates   = (float*)take((size_t)BB * 4 * HH * 4);
  float*    cbuf    = (float*)take((size_t)BB * HH * 4);

  auto cvt = [&](const float* s, _Float16* d, long n) {
    f32_to_f16_kernel<<<dim3((unsigned)((n + 255) / 256)), dim3(256), 0, stream>>>(s, d, n);
  };
  auto pack = [&](const float* s, _Float16* d, int K, int N) {
    int total = (K >> 5) * (N >> 4) * 32;
    pack_b_kernel<<<dim3((total + 255) / 256), dim3(256), 0, stream>>>(s, d, K, N);
  };
  auto gemm = [&](const _Float16* A, int lda, const _Float16* Bp, const float* bias,
                  const float* extra, long ldextra, float* Cf32, _Float16* Cf16,
                  int ldc, int M, int N, int K) {
    int strips = (M >> 5) * (N >> 6);   // 32x64 per wave
    gemm_wmma_kernel<<<dim3((strips + 7) / 8), dim3(256), 0, stream>>>(
        A, lda, Bp, bias, extra, ldextra, Cf32, Cf16, ldc, M, N, K);
  };

  // ---- one-time prep ----
  cvt(enc, encf16, (long)BB * SS * DENC);
  cvt(inputs, xinf16, (long)BB * TT * DIN);
  pack(W1, W1p, DENC, HH);
  pack(W2, W2p, HH, HH);
  pack(Wp, WpTopP, DIN, HH);
  pack(Wp + (size_t)DIN * HH, WpBotP, DENC, HH);
  pack(Wk, WkrP, HH, 4 * HH);
  pack(Wr, WkrP + (size_t)(HH >> 5) * ((4 * HH) >> 4) * 32 * 16, HH, 4 * HH);
  init_state_kernel<<<dim3(BB * HH / 256), dim3(256), 0, stream>>>(h0, c0, zh, cbuf);

  // enc_proj = enc @ W1 + b1  -> f16 [B*S, H]
  gemm(encf16, DENC, W1p, b1, nullptr, 0, nullptr, encproj, HH, BB * SS, HH, DENC);
  // xproj = inputs @ Wp[:DIN]  -> f32 [B*T, H]  (bp added later in z GEMM)
  gemm(xinf16, DIN, WpTopP, nullptr, nullptr, 0, xproj, nullptr, HH, BB * TT, HH, DIN);

  const long scoreWaves = (long)BB * SS;
  const int scoreBlocks = (int)((scoreWaves * 32 + 255) / 256);

  for (int t = 0; t < TT; ++t) {
    // dec_proj = h @ W2 + b2
    gemm(zh + HH, 2 * HH, W2p, b2, nullptr, 0, decproj, nullptr, HH, BB, HH, HH);
    // scores[b,s]
    score_kernel<<<dim3(scoreBlocks), dim3(256), 0, stream>>>(encproj, decproj, V, scores);
    // softmax over S (in place)
    softmax_kernel<<<dim3(BB), dim3(256), 0, stream>>>(scores);
    // context[b,:]
    context_kernel<<<dim3(BB), dim3(256), 0, stream>>>(scores, encf16, ctx);
    // z = ctx @ Wp[DIN:] + xproj[:,t,:] + bp  -> f16 into zh[:, 0:H]
    gemm(ctx, DENC, WpBotP, bp, xproj + (size_t)t * HH, (long)TT * HH,
         nullptr, zh, 2 * HH, BB, HH, DENC);
    // gates = [z|h] @ [Wk;Wr] + bl  -> f32 [B, 4H]
    gemm(zh, 2 * HH, WkrP, bl, nullptr, 0, gates, nullptr, 4 * HH, BB, 4 * HH, 2 * HH);
    // LSTM pointwise; writes h into zh and out[:,t,:]
    lstm_kernel<<<dim3(BB * HH / 256), dim3(256), 0, stream>>>(gates, cbuf, zh, out, t);
  }
}
